// ffMoE_71605694759039
// MI455X (gfx1250) — compile-verified, matrix operand order checked by reading, not run on hardware
//
#include <hip/hip_runtime.h>
#include <hip/hip_bf16.h>
#include <stdint.h>

// Problem constants (b=1, s=2048, h=1024, E=8, f=4096, K=2)
#define T 2048
#define H 1024
#define E 8
#define F 4096
#define KTOP 2

#define TM 128
#define TN 128
#define TK 32

typedef __attribute__((ext_vector_type(16))) __bf16 bfrag;
typedef __attribute__((ext_vector_type(2)))  __bf16 bf16x2;
typedef __attribute__((ext_vector_type(8)))  float v8f;

union FragU { uint4 q[2]; bfrag f; };
union F4U   { float4 v; float f[4]; };

// Load one 16x32 bf16 WMMA operand tile from LDS.
// Layout per ISA 7.12.2 (16-bit A 16x32): lane L -> row = L%16,
// v0..3 hold K = (L/16)*8 + {0..7}, v4..7 hold K = 16 + (L/16)*8 + {0..7}.
__device__ __forceinline__ bfrag load_frag(const __bf16* base, int stride, int lane) {
    int m  = lane & 15;
    int hi = lane >> 4;
    const __bf16* p = base + m * stride + hi * 8;
    FragU u;
    u.q[0] = *(const uint4*)(p);
    u.q[1] = *(const uint4*)(p + 16);
    return u.f;
}

__device__ __forceinline__ unsigned lds_off(const void* p) {
    // generic LDS address low 32 bits == LDS byte offset (ISA 10.2 aperture decode)
    return (unsigned)(size_t)p;
}

// CDNA5 async global->LDS copy (ASYNCcnt-tracked), 16 bytes per lane.
__device__ __forceinline__ void async_copy_b128(unsigned lds_dst, const void* gaddr) {
    asm volatile("global_load_async_to_lds_b128 %0, %1, off"
                 :: "v"(lds_dst), "v"(gaddr) : "memory");
}
__device__ __forceinline__ void wait_async0() {
    asm volatile("s_wait_asynccnt 0x0" ::: "memory");
}

// Two f32 -> packed bf16x2 in one dword; pattern folds to v_cvt_pk_bf16_f32.
__device__ __forceinline__ unsigned pack_bf16(float lo, float hi) {
    bf16x2 p = { (__bf16)lo, (__bf16)hi };
    return __builtin_bit_cast(unsigned, p);
}

// Shared GEMM core: acc += gather(A)[TMxK] * cvt_bf16(B)[KxTN]
// A: bf16 rows gathered via toks[] (pre-clamped to a valid row id).
// B: fp32, row stride Nstride, converted to bf16 in LDS (transposed, K-contiguous).
__device__ __forceinline__ void gemm_core(
    const __bf16* __restrict__ Aglob, size_t arow_stride,
    const float* __restrict__ Bglob, int Kdim, int Nstride,
    const int* toks,
    __bf16 (*As)[TM][TK],           // [2][TM][TK] double buffer
    __bf16 (*Bs)[TK + 8],           // [TN][TK+8]
    int tid, v8f acc[4][2])
{
    const int lane = tid & 31;
    const int warp = tid >> 5;
    const int wm = warp >> 2;       // 0..1
    const int wn = warp & 3;        // 0..3

    // ---- A: async gather, 2 x b128 per thread per K-step --------------------
    const int r0  = tid >> 2;             // 0..63 (second row = r0+64)
    const int seg = (tid & 3) * 8;        // bf16 element offset within row
    const int t0 = toks[r0];
    const int t1 = toks[r0 + 64];
    const __bf16* ga0 = Aglob + (size_t)t0 * arow_stride + seg;
    const __bf16* ga1 = Aglob + (size_t)t1 * arow_stride + seg;
    unsigned la0[2], la1[2];
    la0[0] = lds_off(&As[0][r0][seg]);      la0[1] = lds_off(&As[1][r0][seg]);
    la1[0] = lds_off(&As[0][r0 + 64][seg]); la1[1] = lds_off(&As[1][r0 + 64][seg]);

    // ---- B: register-staged fp32 prefetch, 4 x float4 per thread ------------
    const int kk_a = (tid >> 5) * 2;      // even, 0..14
    const int kk_b = kk_a + 16;
    const int nn   = (tid & 31) * 4;
    const float* bp = Bglob + nn;
    F4U bra0, bra1, brb0, brb1;

    auto issueA = [&](int k0, int b) {
        async_copy_b128(la0[b], ga0 + k0);
        async_copy_b128(la1[b], ga1 + k0);
    };
    auto loadB = [&](int k0) {
        const float* p = bp + (size_t)(k0 + kk_a) * Nstride;
        bra0.v = *(const float4*)(p);
        bra1.v = *(const float4*)(p + Nstride);
        const float* q = bp + (size_t)(k0 + kk_b) * Nstride;
        brb0.v = *(const float4*)(q);
        brb1.v = *(const float4*)(q + Nstride);
    };
    auto storeB = [&]() {
#pragma unroll
        for (int i = 0; i < 4; i++) {
            *(unsigned*)(&Bs[nn + i][kk_a]) = pack_bf16(bra0.f[i], bra1.f[i]);
            *(unsigned*)(&Bs[nn + i][kk_b]) = pack_bf16(brb0.f[i], brb1.f[i]);
        }
    };
    auto compute = [&](int b) {
        bfrag bfv[2];
#pragma unroll
        for (int j = 0; j < 2; j++)
            bfv[j] = load_frag(&Bs[wn * 32 + j * 16][0], TK + 8, lane);
#pragma unroll
        for (int i = 0; i < 4; i++) {
            bfrag af = load_frag(&As[b][wm * 64 + i * 16][0], TK, lane);
#pragma unroll
            for (int j = 0; j < 2; j++)
                acc[i][j] = __builtin_amdgcn_wmma_f32_16x16x32_bf16(
                    false, af, false, bfv[j], (short)0, acc[i][j], false, false);
        }
    };

    const int nIter = Kdim / TK;
    issueA(0, 0);
    loadB(0);
    wait_async0();
    __syncthreads();                 // As[0] visible from all waves

    int cur = 0;
    for (int it = 0; it < nIter; ++it) {
        const int k0 = it * TK;
        storeB();                     // Bs <- regs (prev compute done at last barrier)
        const bool more = (it + 1 < nIter);
        if (more) issueA(k0 + TK, cur ^ 1);
        __syncthreads();              // Bs ready
        if (more) loadB(k0 + TK);     // overlap global B with WMMA
        compute(cur);
        if (more) wait_async0();      // our async A[next] landed
        __syncthreads();              // everyone done with Bs / As[next] visible
        cur ^= 1;
    }
}

// ---------------------------------------------------------------- kernels ---

__global__ void convert_x_kernel(const float* __restrict__ x, unsigned* __restrict__ xb2) {
    int idx = blockIdx.x * blockDim.x + threadIdx.x;
    if (idx < T * H / 2) {
        float2 v = ((const float2*)x)[idx];
        xb2[idx] = pack_bf16(v.x, v.y);
    }
}

__global__ void init_kernel(int* counts, int* am_counts, float* sum_probs) {
    int i = threadIdx.x;
    if (i < E) { counts[i] = 0; am_counts[i] = 0; sum_probs[i] = 0.0f; }
}

// One block per token: scores = x @ Wr + br, softmax, top-2, routing lists.
__global__ __launch_bounds__(256) void router_kernel(
    const float* __restrict__ x, const float* __restrict__ Wr, const float* __restrict__ br,
    int* __restrict__ counts, int* __restrict__ am_counts, float* __restrict__ sum_probs,
    int* __restrict__ perm_tok, int* __restrict__ perm_slot, float* __restrict__ perm_w)
{
    int tok = blockIdx.x;
    int tid = threadIdx.x;
    float acc[E];
#pragma unroll
    for (int e = 0; e < E; e++) acc[e] = 0.0f;

    for (int i = tid; i < H; i += 256) {
        float xv = x[tok * H + i];
        const float4 w0 = *(const float4*)(Wr + i * E);
        const float4 w1 = *(const float4*)(Wr + i * E + 4);
        acc[0] += xv * w0.x; acc[1] += xv * w0.y; acc[2] += xv * w0.z; acc[3] += xv * w0.w;
        acc[4] += xv * w1.x; acc[5] += xv * w1.y; acc[6] += xv * w1.z; acc[7] += xv * w1.w;
    }

    __shared__ float red[256 * E];
#pragma unroll
    for (int e = 0; e < E; e++) red[tid * E + e] = acc[e];
    __syncthreads();
    for (int s = 128; s > 0; s >>= 1) {
        if (tid < s) {
#pragma unroll
            for (int e = 0; e < E; e++) red[tid * E + e] += red[(tid + s) * E + e];
        }
        __syncthreads();
    }

    if (tid == 0) {
        float sc[E], pr[E];
        float m = -1e30f;
#pragma unroll
        for (int e = 0; e < E; e++) { sc[e] = red[e] + br[e]; m = fmaxf(m, sc[e]); }
        float sum = 0.0f;
#pragma unroll
        for (int e = 0; e < E; e++) { pr[e] = __expf(sc[e] - m); sum += pr[e]; }
        float inv = 1.0f / sum;
#pragma unroll
        for (int e = 0; e < E; e++) {
            pr[e] *= inv;
            atomicAdd(&sum_probs[e], pr[e]);
        }
        int e0 = 0; float p0 = pr[0];
#pragma unroll
        for (int e = 1; e < E; e++) if (pr[e] > p0) { p0 = pr[e]; e0 = e; }
        int e1 = -1; float p1 = -1.0f;
#pragma unroll
        for (int e = 0; e < E; e++) if (e != e0 && pr[e] > p1) { p1 = pr[e]; e1 = e; }
        atomicAdd(&am_counts[e0], 1);
        float wn = 1.0f / (p0 + p1);
        int pos0 = atomicAdd(&counts[e0], 1);
        perm_tok[e0 * T + pos0]  = tok;
        perm_slot[e0 * T + pos0] = tok * 2 + 0;
        perm_w[e0 * T + pos0]    = p0 * wn;
        int pos1 = atomicAdd(&counts[e1], 1);
        perm_tok[e1 * T + pos1]  = tok;
        perm_slot[e1 * T + pos1] = tok * 2 + 1;
        perm_w[e1 * T + pos1]    = p1 * wn;
    }
}

// GEMM1: gathered x rows (bf16) @ W1[e] + b1, relu^2 -> h1 (bf16)
__global__ __launch_bounds__(256) void gemm1_kernel(
    const __bf16* __restrict__ xb, const float* __restrict__ W1, const float* __restrict__ b1,
    const int* __restrict__ perm_tok, const int* __restrict__ perm_slot,
    const int* __restrict__ counts, __bf16* __restrict__ h1)
{
    const int e  = blockIdx.z;
    const int ne = counts[e];
    const int row0 = blockIdx.y * TM;
    if (row0 >= ne) return;
    const int n0  = blockIdx.x * TN;
    const int tid  = threadIdx.x;
    const int lane = tid & 31;
    const int warp = tid >> 5;
    const int wm = warp >> 2;
    const int wn = warp & 3;

    __shared__ __align__(16) __bf16 As[2][TM][TK];
    __shared__ __align__(16) __bf16 Bs[TN][TK + 8];
    __shared__ int toks[TM];
    __shared__ int slots[TM];

    for (int r = tid; r < TM; r += 256) {
        int gr = row0 + r;
        toks[r]  = (gr < ne) ? perm_tok[e * T + gr]  : 0;   // clamp: load-safe
        slots[r] = (gr < ne) ? perm_slot[e * T + gr] : 0;
    }
    __syncthreads();

    v8f acc[4][2] = {};
    gemm_core(xb, H, W1 + (size_t)e * H * F + n0, H, F, toks, As, Bs, tid, acc);

    float bias[2];
#pragma unroll
    for (int j = 0; j < 2; j++)
        bias[j] = b1[e * F + n0 + wn * 32 + j * 16 + (lane & 15)];
#pragma unroll
    for (int i = 0; i < 4; i++) {
        int mbase = wm * 64 + i * 16 + (lane >> 4) * 8;
#pragma unroll
        for (int j = 0; j < 2; j++) {
            int n = n0 + wn * 32 + j * 16 + (lane & 15);
#pragma unroll
            for (int r = 0; r < 8; r++) {
                int ml = mbase + r;
                if (row0 + ml < ne) {
                    float v = acc[i][j][r] + bias[j];
                    v = fmaxf(v, 0.0f);
                    v = v * v;
                    h1[(size_t)slots[ml] * F + n] = (__bf16)v;
                }
            }
        }
    }
}

// GEMM2: h1 (bf16) @ W2[e] + b2, scaled by routing weight -> out_pair[slot]
__global__ __launch_bounds__(256) void gemm2_kernel(
    const __bf16* __restrict__ h1, const float* __restrict__ W2, const float* __restrict__ b2,
    const int* __restrict__ perm_slot, const float* __restrict__ perm_w,
    const int* __restrict__ counts, float* __restrict__ outp)
{
    const int e  = blockIdx.z;
    const int ne = counts[e];
    const int row0 = blockIdx.y * TM;
    if (row0 >= ne) return;
    const int n0  = blockIdx.x * TN;
    const int tid  = threadIdx.x;
    const int lane = tid & 31;
    const int warp = tid >> 5;
    const int wm = warp >> 2;
    const int wn = warp & 3;

    __shared__ __align__(16) __bf16 As[2][TM][TK];
    __shared__ __align__(16) __bf16 Bs[TN][TK + 8];
    __shared__ int   slots[TM];
    __shared__ float wgt[TM];

    for (int r = tid; r < TM; r += 256) {
        int gr = row0 + r;
        slots[r] = (gr < ne) ? perm_slot[e * T + gr] : 0;   // clamp: load-safe
        wgt[r]   = (gr < ne) ? perm_w[e * T + gr]    : 0.0f;
    }
    __syncthreads();

    v8f acc[4][2] = {};
    gemm_core(h1, F, W2 + (size_t)e * F * H + n0, F, H, slots, As, Bs, tid, acc);

    float bias[2];
#pragma unroll
    for (int j = 0; j < 2; j++)
        bias[j] = b2[e * H + n0 + wn * 32 + j * 16 + (lane & 15)];
#pragma unroll
    for (int i = 0; i < 4; i++) {
        int mbase = wm * 64 + i * 16 + (lane >> 4) * 8;
#pragma unroll
        for (int j = 0; j < 2; j++) {
            int n = n0 + wn * 32 + j * 16 + (lane & 15);
#pragma unroll
            for (int r = 0; r < 8; r++) {
                int ml = mbase + r;
                if (row0 + ml < ne) {
                    float v = (acc[i][j][r] + bias[j]) * wgt[ml];
                    outp[(size_t)slots[ml] * H + n] = v;
                }
            }
        }
    }
}

__global__ void combine_kernel(const float* __restrict__ outp, float* __restrict__ out) {
    int idx = blockIdx.x * blockDim.x + threadIdx.x;
    if (idx < T * H) {
        int t = idx >> 10;
        int i = idx & (H - 1);
        out[idx] = outp[(size_t)t * 2 * H + i] + outp[(size_t)t * 2 * H + H + i];
    }
}

__global__ void balloss_kernel(const int* __restrict__ am_counts,
                               const float* __restrict__ sum_probs,
                               float* __restrict__ out_scalar) {
    if (threadIdx.x == 0 && blockIdx.x == 0) {
        float loss = 0.0f;
        const float invT = 1.0f / (float)T;
#pragma unroll
        for (int e = 0; e < E; e++)
            loss += ((float)am_counts[e] * invT) * (sum_probs[e] * invT);
        out_scalar[0] = 0.001f * loss;
    }
}

// ------------------------------------------------------------------- host ---

static inline size_t align_up(size_t v, size_t a) { return (v + a - 1) & ~(a - 1); }

extern "C" void kernel_launch(void* const* d_in, const int* in_sizes, int n_in,
                              void* d_out, int out_size, void* d_ws, size_t ws_size,
                              hipStream_t stream) {
    const float* x  = (const float*)d_in[0];
    const float* Wr = (const float*)d_in[1];
    const float* br = (const float*)d_in[2];
    const float* W1 = (const float*)d_in[3];
    const float* b1 = (const float*)d_in[4];
    const float* W2 = (const float*)d_in[5];
    const float* b2 = (const float*)d_in[6];
    float* out = (float*)d_out;   // [T*H] out, then 1 float bal_loss

    char* ws = (char*)d_ws;
    size_t o = 0;
    __bf16* xb       = (__bf16*)(ws + o); o = align_up(o + (size_t)T * H * 2, 256);
    int*   perm_tok  = (int*)  (ws + o); o = align_up(o + (size_t)E * T * 4, 256);
    int*   perm_slot = (int*)  (ws + o); o = align_up(o + (size_t)E * T * 4, 256);
    float* perm_w    = (float*)(ws + o); o = align_up(o + (size_t)E * T * 4, 256);
    int*   counts    = (int*)  (ws + o); o = align_up(o + E * 4, 256);
    int*   am_counts = (int*)  (ws + o); o = align_up(o + E * 4, 256);
    float* sum_probs = (float*)(ws + o); o = align_up(o + E * 4, 256);
    __bf16* h1       = (__bf16*)(ws + o); o = align_up(o + (size_t)T * KTOP * F * 2, 256);
    float* outp      = (float*)(ws + o); o = align_up(o + (size_t)T * KTOP * H * 4, 256);
    (void)ws_size; (void)in_sizes; (void)n_in; (void)out_size;

    convert_x_kernel<<<(T * H / 2 + 255) / 256, 256, 0, stream>>>(x, (unsigned*)xb);
    init_kernel<<<1, 64, 0, stream>>>(counts, am_counts, sum_probs);
    router_kernel<<<T, 256, 0, stream>>>(x, Wr, br, counts, am_counts, sum_probs,
                                         perm_tok, perm_slot, perm_w);
    gemm1_kernel<<<dim3(F / TN, T / TM, E), 256, 0, stream>>>(
        xb, W1, b1, perm_tok, perm_slot, counts, h1);
    gemm2_kernel<<<dim3(H / TN, T / TM, E), 256, 0, stream>>>(
        h1, W2, b2, perm_slot, perm_w, counts, outp);
    combine_kernel<<<(T * H + 255) / 256, 256, 0, stream>>>(outp, out);
    balloss_kernel<<<1, 64, 0, stream>>>(am_counts, sum_probs, out + (size_t)T * H);
}
